// ConvDropoutNormNonlinAttention_14164802143009
// MI455X (gfx1250) — matrix operand order repaired and checked by reading, not compiled, not run on hardware
//
#include <hip/hip_runtime.h>
#include <hip/hip_bf16.h>

// ---------------------------------------------------------------------------
// Problem constants (from reference): B=2, Cin=32, Cout=64, D=H=W=20, N=8000
// ---------------------------------------------------------------------------
#define BATCH   2
#define CIN     32
#define CC      64          // Cout == head dim
#define NTOK    8000        // D*H*W
#define EPS_IN  1e-5f
#define NSLOPE  0.01f

typedef __attribute__((ext_vector_type(16))) __bf16 v16bf;
typedef __attribute__((ext_vector_type(8)))  float  v8f;
typedef __attribute__((ext_vector_type(4)))  unsigned u32x4;

union FragBF {
    v16bf v;
    u32x4 q[2];   // q[0] -> elements 0..7 (VGPR0-3), q[1] -> elements 8..15 (VGPR4-7)
};

// ---------------------------------------------------------------------------
// Kernel 1: direct 3D conv, k=3, SAME padding, + bias.  y[b][o][d][h][w]
// ---------------------------------------------------------------------------
__global__ __launch_bounds__(256) void conv3d_kernel(
    const float* __restrict__ x, const float* __restrict__ w,
    const float* __restrict__ bias, float* __restrict__ y)
{
    int idx = blockIdx.x * blockDim.x + threadIdx.x;
    if (idx >= BATCH * CC * NTOK) return;
    int n  = idx % NTOK;
    int bo = idx / NTOK;
    int o  = bo & (CC - 1);
    int b  = bo >> 6;
    int wi = n % 20;
    int dh = n / 20;
    int h  = dh % 20;
    int d  = dh / 20;

    float acc = bias[o];
    const float* xb = x + (size_t)b * CIN * NTOK;
    const float* wo = w + (size_t)o * CIN * 27;

    for (int c = 0; c < CIN; ++c) {
        const float* xc = xb + c * NTOK;
        const float* wc = wo + c * 27;
        #pragma unroll
        for (int kd = 0; kd < 3; ++kd) {
            int dd = d + kd - 1;
            if ((unsigned)dd >= 20u) continue;
            #pragma unroll
            for (int kh = 0; kh < 3; ++kh) {
                int hh = h + kh - 1;
                if ((unsigned)hh >= 20u) continue;
                #pragma unroll
                for (int kw = 0; kw < 3; ++kw) {
                    int ww = wi + kw - 1;
                    if ((unsigned)ww >= 20u) continue;
                    acc += xc[(dd * 20 + hh) * 20 + ww] * wc[kd * 9 + kh * 3 + kw];
                }
            }
        }
    }
    y[idx] = acc;
}

// ---------------------------------------------------------------------------
// Kernel 2: instance-norm statistics per (b, c):  mu, rstd = rsqrt(var + eps)
// ---------------------------------------------------------------------------
__global__ __launch_bounds__(256) void instnorm_stats_kernel(
    const float* __restrict__ y, float* __restrict__ mu, float* __restrict__ rstd)
{
    int bc = blockIdx.x;
    const float* p = y + (size_t)bc * NTOK;
    float s = 0.f, ss = 0.f;
    for (int i = threadIdx.x; i < NTOK; i += 256) {
        float v = p[i];
        s += v; ss += v * v;
    }
    #pragma unroll
    for (int m = 16; m >= 1; m >>= 1) {
        s  += __shfl_xor(s,  m, 32);
        ss += __shfl_xor(ss, m, 32);
    }
    __shared__ float sh[2][8];
    int wv = threadIdx.x >> 5, ln = threadIdx.x & 31;
    if (ln == 0) { sh[0][wv] = s; sh[1][wv] = ss; }
    __syncthreads();
    if (threadIdx.x == 0) {
        float S = 0.f, SS = 0.f;
        #pragma unroll
        for (int i = 0; i < 8; ++i) { S += sh[0][i]; SS += sh[1][i]; }
        float m_  = S  * (1.0f / NTOK);
        float var = SS * (1.0f / NTOK) - m_ * m_;
        mu[bc]   = m_;
        rstd[bc] = rsqrtf(var + EPS_IN);
    }
}

// ---------------------------------------------------------------------------
// Kernel 3: fused norm + leakyrelu + QKV projections.
//   qT[b][n][c]  (row-major over c)  -> WMMA A fragments, contiguous b128
//   kT[b][m][c]  (row-major over c)  -> WMMA B fragments for S
//   vN[b][c][m]  (natural)           -> WMMA B fragments for O = P x V^T
// ---------------------------------------------------------------------------
__global__ __launch_bounds__(256) void qkv_kernel(
    const float* __restrict__ y, const float* __restrict__ mu,
    const float* __restrict__ rstd,
    const float* __restrict__ gamma, const float* __restrict__ beta,
    const float* __restrict__ wq, const float* __restrict__ bq,
    const float* __restrict__ wk, const float* __restrict__ bk,
    const float* __restrict__ wv, const float* __restrict__ bv,
    __bf16* __restrict__ qT, __bf16* __restrict__ kT, __bf16* __restrict__ vN)
{
    __shared__ float f[CC][128];
    const int NT  = (NTOK + 127) / 128;         // 63 tiles per batch
    int blk = blockIdx.x;
    int b   = blk / NT;
    int n0  = (blk - b * NT) * 128;

    // Stage normalized + activated features for this n-tile in LDS.
    for (int i = threadIdx.x; i < CC * 128; i += 256) {
        int c  = i >> 7;
        int nl = i & 127;
        int n  = n0 + nl;
        float val = 0.f;
        if (n < NTOK) {
            int bc = b * CC + c;
            float v = y[(size_t)bc * NTOK + n];
            v = (v - mu[bc]) * rstd[bc] * gamma[c] + beta[c];
            val = (v >= 0.f) ? v : NSLOPE * v;
        }
        f[c][nl] = val;
    }
    __syncthreads();

    int nl = threadIdx.x & 127;
    int og = (threadIdx.x >> 7) * 32;           // o in [og, og+32)
    int n  = n0 + nl;
    if (n >= NTOK) return;

    for (int oo = 0; oo < 32; ++oo) {
        int o = og + oo;
        const float* wqo = wq + o * CC;
        const float* wko = wk + o * CC;
        const float* wvo = wv + o * CC;
        float aq = bq[o], ak = bk[o], av = bv[o];
        #pragma unroll 4
        for (int c = 0; c < CC; ++c) {
            float fv = f[c][nl];
            aq += wqo[c] * fv;
            ak += wko[c] * fv;
            av += wvo[c] * fv;
        }
        size_t nb = (size_t)b * NTOK + n;
        qT[nb * CC + o] = (__bf16)aq;
        kT[nb * CC + o] = (__bf16)ak;
        vN[((size_t)b * CC + o) * NTOK + n] = (__bf16)av;
    }
}

// ---------------------------------------------------------------------------
// Kernel 4: flash attention with bf16 WMMA, f32 accumulation.
//   One wave = one 16-row query tile; KEY TILE = 64 (amortizes the per-row
//   shfl-butterfly softmax cost over 16 WMMAs instead of 8).
//   S = Q^T K  : 8x v_wmma_f32_16x16x32_bf16 per iter (4 col-tiles x 2 K-steps)
//   O += P V^T : 8x v_wmma_f32_16x16x32_bf16 per iter (4 c-tiles  x 2 K-steps)
// ---------------------------------------------------------------------------
#define PSTR 72   // P-tile row stride in elements: 144B pitch, 16B aligned,
                  // 36-bank row step -> conflict-free for 16 rows.

__global__ __launch_bounds__(128) void flash_attn_kernel(
    const __bf16* __restrict__ qT, const __bf16* __restrict__ kT,
    const __bf16* __restrict__ vN, float* __restrict__ out)
{
    // Per-wave P transpose buffer: 16 rows x 64 cols, padded stride PSTR.
    __shared__ __align__(16) __bf16 pbuf[4][16 * PSTR];

    const int wave = threadIdx.x >> 5;
    const int lane = threadIdx.x & 31;
    const int r    = lane & 15;
    const int hf   = lane >> 4;

    const int QT_PER_B = NTOK / 16;             // 500
    int qt = blockIdx.x * 4 + wave;
    int b  = qt / QT_PER_B;
    int n0 = (qt - b * QT_PER_B) * 16;

    const __bf16* qTb = qT + (size_t)b * NTOK * CC;
    const __bf16* kTb = kT + (size_t)b * NTOK * CC;
    const __bf16* vb  = vN + (size_t)b * CC * NTOK;

    // Q fragments for K-steps c=0..31 and c=32..63 (ISA 16-bit A 16x32 layout:
    // lanes 0-15: VGPR0-3 K=0..7, VGPR4-7 K=16..23; lanes 16-31 shifted by 8).
    FragBF aq0, aq1;
    {
        const __bf16* base = qTb + (size_t)(n0 + r) * CC;
        aq0.q[0] = *(const u32x4*)(base +      hf * 8);
        aq0.q[1] = *(const u32x4*)(base + 16 + hf * 8);
        aq1.q[0] = *(const u32x4*)(base + 32 + hf * 8);
        aq1.q[1] = *(const u32x4*)(base + 48 + hf * 8);
    }

    v8f O0 = {}, O1 = {}, O2 = {}, O3 = {};
    float mrow[8], lrow[8];
    #pragma unroll
    for (int i = 0; i < 8; ++i) { mrow[i] = -3.0e38f; lrow[i] = 0.f; }

    __bf16* p = pbuf[wave];

    for (int m0 = 0; m0 < NTOK; m0 += 64) {
        // ---- S = Q^T K for four 16-col key tiles -------------------------
        // B 32x16 layout (matches A): lanes 0-15 K=0..7,16..23; lanes 16-31 K=8..15,24..31
        // -> two 8-bf16 chunks at element offsets hf*8 and 16+hf*8 within a kT row.
        v8f s0 = {}, s1 = {}, s2 = {}, s3 = {};
        FragBF bfr;
        #define S_TILE(SREG, TOFF)                                              \
        {                                                                       \
            const __bf16* kb = kTb + (size_t)(m0 + (TOFF) + r) * CC;            \
            bfr.q[0] = *(const u32x4*)(kb + hf * 8);                            \
            bfr.q[1] = *(const u32x4*)(kb + 16 + hf * 8);                       \
            SREG = __builtin_amdgcn_wmma_f32_16x16x32_bf16(false, aq0.v, false, \
                        bfr.v, (short)0, SREG, false, false);                   \
            bfr.q[0] = *(const u32x4*)(kb + 32 + hf * 8);                       \
            bfr.q[1] = *(const u32x4*)(kb + 48 + hf * 8);                       \
            SREG = __builtin_amdgcn_wmma_f32_16x16x32_bf16(false, aq1.v, false, \
                        bfr.v, (short)0, SREG, false, false);                   \
        }
        S_TILE(s0, 0)
        S_TILE(s1, 16)
        S_TILE(s2, 32)
        S_TILE(s3, 48)
        #undef S_TILE

        // ---- online softmax.  D layout: VGPR i <-> row i + hf*8; a row's 16
        // columns live across one 16-lane half-wave -> shfl_xor reductions.
        float alpha[8];
        #pragma unroll
        for (int i = 0; i < 8; ++i) {
            float v0 = s0[i], v1 = s1[i], v2 = s2[i], v3 = s3[i];
            float mx = fmaxf(fmaxf(v0, v1), fmaxf(v2, v3));
            mx = fmaxf(mx, __shfl_xor(mx, 1, 32));
            mx = fmaxf(mx, __shfl_xor(mx, 2, 32));
            mx = fmaxf(mx, __shfl_xor(mx, 4, 32));
            mx = fmaxf(mx, __shfl_xor(mx, 8, 32));
            float mnew = fmaxf(mrow[i], mx);
            float p0 = __expf(v0 - mnew);
            float p1 = __expf(v1 - mnew);
            float p2 = __expf(v2 - mnew);
            float p3 = __expf(v3 - mnew);
            float rs = (p0 + p1) + (p2 + p3);
            rs += __shfl_xor(rs, 1, 32);
            rs += __shfl_xor(rs, 2, 32);
            rs += __shfl_xor(rs, 4, 32);
            rs += __shfl_xor(rs, 8, 32);
            float a = __expf(mrow[i] - mnew);
            lrow[i] = lrow[i] * a + rs;
            mrow[i] = mnew;
            alpha[i] = a;
            int row = i + hf * 8;
            p[row * PSTR +      r] = (__bf16)p0;   // key cols  0..15
            p[row * PSTR + 16 + r] = (__bf16)p1;   // key cols 16..31
            p[row * PSTR + 32 + r] = (__bf16)p2;   // key cols 32..47
            p[row * PSTR + 48 + r] = (__bf16)p3;   // key cols 48..63
        }
        #pragma unroll
        for (int i = 0; i < 8; ++i) {
            O0[i] *= alpha[i]; O1[i] *= alpha[i];
            O2[i] *= alpha[i]; O3[i] *= alpha[i];
        }

        // ---- transpose P (D layout -> A layout) via per-wave LDS tile -----
        FragBF ap0, ap1;
        const __bf16* pb = p + r * PSTR;
        ap0.q[0] = *(const u32x4*)(pb +      hf * 8);   // K(m)  0..31
        ap0.q[1] = *(const u32x4*)(pb + 16 + hf * 8);
        ap1.q[0] = *(const u32x4*)(pb + 32 + hf * 8);   // K(m) 32..63
        ap1.q[1] = *(const u32x4*)(pb + 48 + hf * 8);

        // ---- O += P x V^T over 4 c-tiles x 2 K-steps ---------------------
        FragBF vf;
        #define O_TILE(OREG, CT)                                                \
        {                                                                       \
            const __bf16* vp = vb + (size_t)((CT) * 16 + r) * NTOK + m0;        \
            vf.q[0] = *(const u32x4*)(vp + hf * 8);                             \
            vf.q[1] = *(const u32x4*)(vp + 16 + hf * 8);                        \
            OREG = __builtin_amdgcn_wmma_f32_16x16x32_bf16(false, ap0.v, false, \
                        vf.v, (short)0, OREG, false, false);                    \
            vf.q[0] = *(const u32x4*)(vp + 32 + hf * 8);                        \
            vf.q[1] = *(const u32x4*)(vp + 48 + hf * 8);                        \
            OREG = __builtin_amdgcn_wmma_f32_16x16x32_bf16(false, ap1.v, false, \
                        vf.v, (short)0, OREG, false, false);                    \
        }
        O_TILE(O0, 0)
        O_TILE(O1, 1)
        O_TILE(O2, 2)
        O_TILE(O3, 3)
        #undef O_TILE
    }

    // ---- epilogue: out[b][c][n] = O / l ----------------------------------
    #pragma unroll
    for (int i = 0; i < 8; ++i) {
        float inv = 1.0f / lrow[i];
        int n = n0 + i + hf * 8;
        out[((size_t)b * CC + 0 * 16 + r) * NTOK + n] = O0[i] * inv;
        out[((size_t)b * CC + 1 * 16 + r) * NTOK + n] = O1[i] * inv;
        out[((size_t)b * CC + 2 * 16 + r) * NTOK + n] = O2[i] * inv;
        out[((size_t)b * CC + 3 * 16 + r) * NTOK + n] = O3[i] * inv;
    }
}

// ---------------------------------------------------------------------------
// Launcher
// ---------------------------------------------------------------------------
extern "C" void kernel_launch(void* const* d_in, const int* in_sizes, int n_in,
                              void* d_out, int out_size, void* d_ws, size_t ws_size,
                              hipStream_t stream)
{
    (void)in_sizes; (void)n_in; (void)out_size; (void)ws_size;

    const float* x      = (const float*)d_in[0];
    const float* conv_w = (const float*)d_in[1];
    const float* conv_b = (const float*)d_in[2];
    const float* gamma  = (const float*)d_in[3];
    const float* beta   = (const float*)d_in[4];
    const float* wq     = (const float*)d_in[5];
    const float* bq     = (const float*)d_in[6];
    const float* wk     = (const float*)d_in[7];
    const float* bk     = (const float*)d_in[8];
    const float* wv     = (const float*)d_in[9];
    const float* bv     = (const float*)d_in[10];

    // Workspace layout (≈10.3 MB, all L2-resident on MI455X):
    char* ws = (char*)d_ws;
    float*  y    = (float*)ws;                              //  4,096,000 B
    float*  mu   = (float*)(ws + 4096000);                  //        512 B
    float*  rstd = mu + 128;                                //        512 B
    __bf16* qT   = (__bf16*)(ws + 4096000 + 2048);          //  2,048,000 B
    __bf16* kT   = qT + (size_t)BATCH * NTOK * CC;          //  2,048,000 B
    __bf16* vN   = kT + (size_t)BATCH * NTOK * CC;          //  2,048,000 B

    conv3d_kernel<<<(BATCH * CC * NTOK + 255) / 256, 256, 0, stream>>>(
        x, conv_w, conv_b, y);

    instnorm_stats_kernel<<<BATCH * CC, 256, 0, stream>>>(y, mu, rstd);

    qkv_kernel<<<BATCH * ((NTOK + 127) / 128), 256, 0, stream>>>(
        y, mu, rstd, gamma, beta, wq, bq, wk, bk, wv, bv, qT, kT, vN);

    // 2 batches * 500 query tiles / 4 waves per block = 250 blocks
    flash_attn_kernel<<<250, 128, 0, stream>>>(qT, kT, vN, (float*)d_out);
}
